// DECODERLAYER_31361851195402
// MI455X (gfx1250) — matrix operand order repaired and checked by reading, not compiled, not run on hardware
//
#include <hip/hip_runtime.h>
#include <math.h>

// ---------------------------------------------------------------------------
// Types for gfx1250 WMMA (wave32)
// ---------------------------------------------------------------------------
typedef __bf16 bf16;
typedef __attribute__((ext_vector_type(16))) __bf16 v16bf;
typedef __attribute__((ext_vector_type(8)))  __bf16 v8bf;
typedef __attribute__((ext_vector_type(8)))  float  v8f;

#define EMBED 1024
#define HEADS 16
#define DHEAD 64
#define SEQ   2048
#define BATCH 2
#define MROWS (BATCH * SEQ)   // 4096 token rows
#define FFN   (4 * EMBED)     // 4096

// D = A(16x32 bf16) * B(32x16 bf16) + C(16x16 f32)
__device__ __forceinline__ v8f wmma_bf(v16bf a, v16bf b, v8f c) {
  return __builtin_amdgcn_wmma_f32_16x16x32_bf16(false, a, false, b, (short)0, c,
                                                 false, false);
}

// ---------------------------------------------------------------------------
// Fragment loaders (layouts per CDNA5 ISA 7.12.2, wave32)
// A 16x32 (row-major source): lane r=lane&15 holds row r;
//   elems 0..7  = K (half*8 + i), elems 8..15 = K (16 + half*8 + i)
// ---------------------------------------------------------------------------
__device__ __forceinline__ v16bf load_a_frag(const bf16* __restrict__ A, int lda,
                                             int row0, int k0, int lane) {
  const int half = (lane >> 4) & 1;
  const int r    = lane & 15;
  const bf16* p = A + (size_t)(row0 + r) * lda + k0 + half * 8;
  v8bf lo = *reinterpret_cast<const v8bf*>(p);
  v8bf hi = *reinterpret_cast<const v8bf*>(p + 16);
  v16bf o;
#pragma unroll
  for (int i = 0; i < 8; ++i) { o[i] = lo[i]; o[8 + i] = hi[i]; }
  return o;
}

// B 32x16 where B[k][n] = W[n0+n][k0+k] (W row-major [N,K]):
//   lane n=lane&15 holds column n; elems 0..7 = K (half*8 + i),
//   elems 8..15 = K (16 + half*8 + i) -> two 16B chunks from one W row.
__device__ __forceinline__ v16bf load_bt_frag(const bf16* __restrict__ W, int ldw,
                                              int n0, int k0, int lane) {
  const int half = (lane >> 4) & 1;
  const int n    = lane & 15;
  const bf16* p = W + (size_t)(n0 + n) * ldw + k0 + half * 8;
  v8bf lo = *reinterpret_cast<const v8bf*>(p);
  v8bf hi = *reinterpret_cast<const v8bf*>(p + 16);
  v16bf o;
#pragma unroll
  for (int i = 0; i < 8; ++i) { o[i] = lo[i]; o[8 + i] = hi[i]; }
  return o;
}

// ---------------------------------------------------------------------------
// Positional encoding + residual copy + bf16 operand
// ---------------------------------------------------------------------------
__launch_bounds__(256)
__global__ void pe_add_kernel(const float* __restrict__ in, float* __restrict__ xf,
                              bf16* __restrict__ xb) {
  const size_t idx = (size_t)blockIdx.x * 256 + threadIdx.x;  // over B*S*E
  const int e = (int)(idx % EMBED);
  const int s = (int)((idx / EMBED) % SEQ);
  const float fe = (float)(e & ~1);
  const float freq = __powf(10000.0f, -fe * (1.0f / (float)EMBED));
  const float t = (float)s * freq;
  const float arg = (e & 1) ? __cosf(t) : t;
  const float v = in[idx] + __sinf(arg);
  xf[idx] = v;
  xb[idx] = (bf16)v;
}

__launch_bounds__(256)
__global__ void f2b_kernel(const float* __restrict__ in, bf16* __restrict__ out,
                           int n) {
  const int i = blockIdx.x * 256 + threadIdx.x;
  if (i < n) out[i] = (bf16)in[i];
}

// Encoder V: f32 [B,S,H,Dh] -> bf16 [B,H,Dh,S] (output-coalesced transpose)
__launch_bounds__(256)
__global__ void f2b_transpose_v(const float* __restrict__ in, bf16* __restrict__ out) {
  const size_t idx = (size_t)blockIdx.x * 256 + threadIdx.x;  // over B*H*Dh*S
  const int s = (int)(idx & (SEQ - 1));
  const int d = (int)((idx >> 11) & (DHEAD - 1));
  const int h = (int)((idx >> 17) & (HEADS - 1));
  const int b = (int)(idx >> 21);
  const size_t src = ((size_t)(b * SEQ + s) * HEADS + h) * DHEAD + d;
  out[idx] = (bf16)in[src];
}

// ---------------------------------------------------------------------------
// GEMM: C[M,N] = A[M,K](bf16) * W[N,K]^T(bf16) + bias, with epilogue
// EPI 0: bias, *oscale -> bf16   EPI 1: bias + exact GELU -> bf16
// EPI 2: bias -> f32             EPI 3: bias, *oscale -> bf16 [B,H,Dh,S]
// Block = 256 threads = 8 waves, arranged 2(M) x 4(N); wave tile 64x32
// (4 M-subtiles x 2 N-subtiles = 8 WMMAs per 32-wide k-step).
// ---------------------------------------------------------------------------
template <int EPI>
__launch_bounds__(256)
__global__ void gemm_bf16_wmma(const bf16* __restrict__ A, const bf16* __restrict__ W,
                               const float* __restrict__ bias,
                               bf16* __restrict__ outb, float* __restrict__ outf,
                               int M, int N, int K, float oscale) {
  const int lane = threadIdx.x & 31;
  const int wv   = threadIdx.x >> 5;
  const int m0 = blockIdx.x * 128 + (wv >> 2) * 64;
  const int n0 = blockIdx.y * 128 + (wv & 3) * 32;

  v8f acc[4][2] = {};
#pragma unroll 2
  for (int k = 0; k < K; k += 32) {
    if (k + 32 < K) {  // pull next k-slab toward the caches (global_prefetch)
      __builtin_prefetch(A + (size_t)(m0 + lane) * K + k + 32, 0, 3);
      __builtin_prefetch(A + (size_t)(m0 + 32 + lane) * K + k + 32, 0, 3);
      __builtin_prefetch(W + (size_t)(n0 + lane) * K + k + 32, 0, 3);
    }
    v16bf b0 = load_bt_frag(W, K, n0,      k, lane);
    v16bf b1 = load_bt_frag(W, K, n0 + 16, k, lane);
#pragma unroll
    for (int mi = 0; mi < 4; ++mi) {
      v16bf a = load_a_frag(A, K, m0 + mi * 16, k, lane);
      acc[mi][0] = wmma_bf(a, b0, acc[mi][0]);
      acc[mi][1] = wmma_bf(a, b1, acc[mi][1]);
    }
  }

  const int half = lane >> 4, col = lane & 15;
#pragma unroll
  for (int mi = 0; mi < 4; ++mi) {
#pragma unroll
    for (int ni = 0; ni < 2; ++ni) {
      const int cn = n0 + ni * 16 + col;
      const float bv = bias[cn];
#pragma unroll
      for (int j = 0; j < 8; ++j) {
        const int rm = m0 + mi * 16 + j + 8 * half;  // C layout: row j+8*half
        float v = acc[mi][ni][j] + bv;
        if (EPI == 0 || EPI == 3) v *= oscale;  // e.g. fold 1/sqrt(Dh) into Q
        if (EPI == 1) v = 0.5f * v * (1.0f + erff(v * 0.70710678118654752f));
        if (EPI == 2) {
          outf[(size_t)rm * N + cn] = v;
        } else if (EPI == 3) {
          // token rm = (b, s); feature cn = (h, d); out layout [B,H,Dh,S]
          const int b = rm >> 11, s = rm & (SEQ - 1);
          const int h = cn >> 6,  d = cn & (DHEAD - 1);
          outb[(((size_t)(b * HEADS + h)) * DHEAD + d) * SEQ + s] = (bf16)v;
        } else {
          outb[(size_t)rm * N + cn] = (bf16)v;
        }
      }
    }
  }
}

// ---------------------------------------------------------------------------
// One 32-wide K tile of flash attention for a 16-row Q tile.
// MASKED=false for interior tiles (kb+31 < q0): no causal compares at all.
// Scores arrive pre-scaled (1/sqrt(Dh) folded into Q).
// ---------------------------------------------------------------------------
template <bool MASKED>
__device__ __forceinline__ void attn_tile(
    int kb, int q0, int lane, const bf16* __restrict__ Kbase, int ld,
    const bf16* __restrict__ Vbase,
    const v16bf& qf0, const v16bf& qf1,
    v8f& acc0, v8f& acc1, v8f& acc2, v8f& acc3,
    float (&mrow)[8], float (&lrow)[8], bf16 (*Pl)[40]) {
  const int half = lane >> 4, col = lane & 15;

  // S = Q K^T : two 16x16 score tiles, Dh=64 -> two k-steps each
  v8f s0 = {}, s1 = {};
  s0 = wmma_bf(qf0, load_bt_frag(Kbase, ld, kb,      0,  lane), s0);
  s0 = wmma_bf(qf1, load_bt_frag(Kbase, ld, kb,      32, lane), s0);
  s1 = wmma_bf(qf0, load_bt_frag(Kbase, ld, kb + 16, 0,  lane), s1);
  s1 = wmma_bf(qf1, load_bt_frag(Kbase, ld, kb + 16, 32, lane), s1);

  // online softmax per row (row j+8*half lives entirely in one 16-lane half)
#pragma unroll
  for (int j = 0; j < 8; ++j) {
    float v0 = s0[j];
    float v1 = s1[j];
    if (MASKED) {
      const int row = q0 + j + 8 * half;
      if (kb + col      > row) v0 = -1e30f;
      if (kb + 16 + col > row) v1 = -1e30f;
    }
    float mt = fmaxf(v0, v1);
#pragma unroll
    for (int o = 8; o > 0; o >>= 1) mt = fmaxf(mt, __shfl_xor(mt, o, 32));
    const float mnew = fmaxf(mrow[j], mt);
    const float sc = __expf(mrow[j] - mnew);
    const float p0 = __expf(v0 - mnew);
    const float p1 = __expf(v1 - mnew);
    float ls = p0 + p1;
#pragma unroll
    for (int o = 8; o > 0; o >>= 1) ls += __shfl_xor(ls, o, 32);
    lrow[j] = lrow[j] * sc + ls;
    mrow[j] = mnew;
    acc0[j] *= sc; acc1[j] *= sc; acc2[j] *= sc; acc3[j] *= sc;
    Pl[j + 8 * half][col]      = (bf16)p0;
    Pl[j + 8 * half][16 + col] = (bf16)p1;
  }

  // Reload P in A-fragment layout (same-wave DS ops are ordered)
  v16bf pf;
  {
    const bf16* pr = &Pl[col][0];
    v8bf lo = *reinterpret_cast<const v8bf*>(pr + half * 8);
    v8bf hi = *reinterpret_cast<const v8bf*>(pr + 16 + half * 8);
#pragma unroll
    for (int i = 0; i < 8; ++i) { pf[i] = lo[i]; pf[8 + i] = hi[i]; }
  }

  // O += P (16x32) * V (32x64): 4 WMMAs; Vt gives contiguous B-fragments
  acc0 = wmma_bf(pf, load_bt_frag(Vbase, SEQ, 0,  kb, lane), acc0);
  acc1 = wmma_bf(pf, load_bt_frag(Vbase, SEQ, 16, kb, lane), acc1);
  acc2 = wmma_bf(pf, load_bt_frag(Vbase, SEQ, 32, kb, lane), acc2);
  acc3 = wmma_bf(pf, load_bt_frag(Vbase, SEQ, 48, kb, lane), acc3);
}

// ---------------------------------------------------------------------------
// Causal attention (flash style), one wave per 16-row Q tile, K tiled by 32.
// Q/K layout: [B, S, H*Dh] flat; V pre-transposed to [B, H, Dh, S] so the PV
// B-fragment is two 16B/lane loads from one Vt row. Block = 128 threads = 4 waves;
// wave-private LDS for the P reshape (no cross-wave sync needed).
// ---------------------------------------------------------------------------
__launch_bounds__(128)
__global__ void attn_causal_wmma(const bf16* __restrict__ Q, const bf16* __restrict__ Km,
                                 const bf16* __restrict__ Vt, float* __restrict__ O) {
  const int lane = threadIdx.x & 31;
  const int wv   = threadIdx.x >> 5;  // 0..3
  const int b = blockIdx.x / HEADS;
  const int h = blockIdx.x % HEADS;
  const int q0 = (blockIdx.y * 4 + wv) * 16;
  const int ld = EMBED;
  const size_t base  = (size_t)b * SEQ * ld + (size_t)h * DHEAD;       // Q/K/O
  const size_t vbase = ((size_t)(b * HEADS + h)) * DHEAD * SEQ;        // Vt

  __shared__ __align__(16) bf16 Pl[4][16][40];  // per-wave 16x32 P tile (+pad)

  const v16bf qf0 = load_a_frag(Q + base, ld, q0, 0,  lane);
  const v16bf qf1 = load_a_frag(Q + base, ld, q0, 32, lane);

  v8f acc0 = {}, acc1 = {}, acc2 = {}, acc3 = {};
  float mrow[8], lrow[8];
#pragma unroll
  for (int j = 0; j < 8; ++j) { mrow[j] = -1e30f; lrow[j] = 0.0f; }

  const int nkt = q0 / 32 + 1;  // causal: K tiles covering keys 0..q0+15

  // interior tiles: entirely below the diagonal, no mask compares
  for (int kt = 0; kt < nkt - 1; ++kt)
    attn_tile<false>(kt * 32, q0, lane, Km + base, ld, Vt + vbase,
                     qf0, qf1, acc0, acc1, acc2, acc3, mrow, lrow, Pl[wv]);
  // diagonal tile: causal mask applied
  attn_tile<true>((nkt - 1) * 32, q0, lane, Km + base, ld, Vt + vbase,
                  qf0, qf1, acc0, acc1, acc2, acc3, mrow, lrow, Pl[wv]);

  const int half = lane >> 4, col = lane & 15;
#pragma unroll
  for (int nb = 0; nb < 4; ++nb) {
    v8f av = (nb == 0) ? acc0 : (nb == 1) ? acc1 : (nb == 2) ? acc2 : acc3;
#pragma unroll
    for (int j = 0; j < 8; ++j) {
      const int row = q0 + j + 8 * half;
      O[base + (size_t)row * ld + nb * 16 + col] = av[j] / lrow[j];
    }
  }
}

// ---------------------------------------------------------------------------
// out = LayerNorm(A + R) * g + be ; one 256-thread block per token row (E=1024)
// ---------------------------------------------------------------------------
__launch_bounds__(256)
__global__ void add_layernorm(const float* __restrict__ A, const float* __restrict__ R,
                              const float* __restrict__ g, const float* __restrict__ be,
                              float* __restrict__ outf, bf16* __restrict__ outb) {
  const size_t off = (size_t)blockIdx.x * EMBED;
  __shared__ float red[8];
  float loc[4];
  float s = 0.0f;
#pragma unroll
  for (int i = 0; i < 4; ++i) {
    const int c = threadIdx.x + i * 256;
    const float v = A[off + c] + R[off + c];
    loc[i] = v; s += v;
  }
#pragma unroll
  for (int o = 16; o > 0; o >>= 1) s += __shfl_xor(s, o, 32);
  if ((threadIdx.x & 31) == 0) red[threadIdx.x >> 5] = s;
  __syncthreads();
  float tot = 0.0f;
#pragma unroll
  for (int i = 0; i < 8; ++i) tot += red[i];
  const float mu = tot * (1.0f / (float)EMBED);
  __syncthreads();
  float vs = 0.0f;
#pragma unroll
  for (int i = 0; i < 4; ++i) { const float d = loc[i] - mu; vs += d * d; }
#pragma unroll
  for (int o = 16; o > 0; o >>= 1) vs += __shfl_xor(vs, o, 32);
  if ((threadIdx.x & 31) == 0) red[threadIdx.x >> 5] = vs;
  __syncthreads();
  float vtot = 0.0f;
#pragma unroll
  for (int i = 0; i < 8; ++i) vtot += red[i];
  const float inv = rsqrtf(vtot * (1.0f / (float)EMBED) + 1e-5f);
#pragma unroll
  for (int i = 0; i < 4; ++i) {
    const int c = threadIdx.x + i * 256;
    const float y = (loc[i] - mu) * inv * g[c] + be[c];
    outf[off + c] = y;
    if (outb) outb[off + c] = (bf16)y;
  }
}

// ---------------------------------------------------------------------------
// Host-side orchestration
// ---------------------------------------------------------------------------
extern "C" void kernel_launch(void* const* d_in, const int* in_sizes, int n_in,
                              void* d_out, int out_size, void* d_ws, size_t ws_size,
                              hipStream_t stream) {
  (void)in_sizes; (void)n_in; (void)out_size; (void)ws_size;

  const float* inp  = (const float*)d_in[0];
  const float* kenc = (const float*)d_in[1];
  const float* venc = (const float*)d_in[2];
  const float* Wqm  = (const float*)d_in[3];  const float* bqm = (const float*)d_in[4];
  const float* Wkm  = (const float*)d_in[5];  const float* bkm = (const float*)d_in[6];
  const float* Wvm  = (const float*)d_in[7];  const float* bvm = (const float*)d_in[8];
  const float* Wqa  = (const float*)d_in[9];  const float* bqa = (const float*)d_in[10];
  const float* W1   = (const float*)d_in[11]; const float* b1  = (const float*)d_in[12];
  const float* W2   = (const float*)d_in[13]; const float* b2  = (const float*)d_in[14];
  const float* g1   = (const float*)d_in[15]; const float* be1 = (const float*)d_in[16];
  const float* g2   = (const float*)d_in[17]; const float* be2 = (const float*)d_in[18];
  const float* g3   = (const float*)d_in[19]; const float* be3 = (const float*)d_in[20];

  char* wsB = (char*)d_ws;
  size_t off = 0;
  auto alloc = [&](size_t bytes) -> void* {
    void* p = wsB + off;
    off = (off + bytes + 255) & ~(size_t)255;
    return p;
  };
  const size_t TE = (size_t)MROWS * EMBED;  // 4M token-elements
  float* x_f   = (float*)alloc(TE * 4);
  float* y1_f  = (float*)alloc(TE * 4);
  float* y2_f  = (float*)alloc(TE * 4);
  float* tmp_f = (float*)alloc(TE * 4);     // ctx1 -> ctx2 -> mlp (reused)
  bf16* x_b  = (bf16*)alloc(TE * 2);
  bf16* y1_b = (bf16*)alloc(TE * 2);
  bf16* y2_b = (bf16*)alloc(TE * 2);
  bf16* Qb   = (bf16*)alloc(TE * 2);        // reused as Qd after attn1
  bf16* Kb   = (bf16*)alloc(TE * 2);
  bf16* Vtb  = (bf16*)alloc(TE * 2);        // self-attn V, [B,H,Dh,S]
  bf16* Keb  = (bf16*)alloc(TE * 2);
  bf16* Vetb = (bf16*)alloc(TE * 2);        // encoder V, [B,H,Dh,S]
  bf16* Wqb  = (bf16*)alloc((size_t)EMBED * EMBED * 2);
  bf16* Wkb  = (bf16*)alloc((size_t)EMBED * EMBED * 2);
  bf16* Wvb  = (bf16*)alloc((size_t)EMBED * EMBED * 2);
  bf16* Wqab = (bf16*)alloc((size_t)EMBED * EMBED * 2);
  bf16* W1b  = (bf16*)alloc((size_t)FFN * EMBED * 2);
  bf16* W2b  = (bf16*)alloc((size_t)EMBED * FFN * 2);
  bf16* h_b  = (bf16*)alloc((size_t)MROWS * FFN * 2);

  const int EE = EMBED * EMBED;       // 1,048,576
  const int FE = FFN * EMBED;         // 4,194,304
  const int TEi = (int)TE;            // 4,194,304
  const float QSCALE = 0.125f;        // 1/sqrt(Dh)

  // 1) x = inputs + PE
  pe_add_kernel<<<(TEi + 255) / 256, 256, 0, stream>>>(inp, x_f, x_b);

  // 2) weight / encoder conversions to bf16 (deterministic, every call)
  f2b_kernel<<<(EE + 255) / 256, 256, 0, stream>>>(Wqm, Wqb, EE);
  f2b_kernel<<<(EE + 255) / 256, 256, 0, stream>>>(Wkm, Wkb, EE);
  f2b_kernel<<<(EE + 255) / 256, 256, 0, stream>>>(Wvm, Wvb, EE);
  f2b_kernel<<<(EE + 255) / 256, 256, 0, stream>>>(Wqa, Wqab, EE);
  f2b_kernel<<<(FE + 255) / 256, 256, 0, stream>>>(W1, W1b, FE);
  f2b_kernel<<<(FE + 255) / 256, 256, 0, stream>>>(W2, W2b, FE);
  f2b_kernel<<<(TEi + 255) / 256, 256, 0, stream>>>(kenc, Keb, TEi);
  f2b_transpose_v<<<(TEi + 255) / 256, 256, 0, stream>>>(venc, Vetb);

  // 3) Q/K/V projections (M=4096, N=1024, K=1024); Q pre-scaled, V transposed
  dim3 gN1024(MROWS / 128, EMBED / 128);
  gemm_bf16_wmma<0><<<gN1024, 256, 0, stream>>>(x_b, Wqb, bqm, Qb, nullptr, MROWS, EMBED, EMBED, QSCALE);
  gemm_bf16_wmma<0><<<gN1024, 256, 0, stream>>>(x_b, Wkb, bkm, Kb, nullptr, MROWS, EMBED, EMBED, 1.0f);
  gemm_bf16_wmma<3><<<gN1024, 256, 0, stream>>>(x_b, Wvb, bvm, Vtb, nullptr, MROWS, EMBED, EMBED, 1.0f);

  // 4) causal self-attention -> ctx (tmp_f)
  dim3 gAttn(BATCH * HEADS, SEQ / 64);
  attn_causal_wmma<<<gAttn, 128, 0, stream>>>(Qb, Kb, Vtb, tmp_f);

  // 5) y1 = LN(ctx + x)
  add_layernorm<<<MROWS, 256, 0, stream>>>(tmp_f, x_f, g1, be1, y1_f, y1_b);

  // 6) Qd projection (reuse Qb, pre-scaled), then cross-attention vs encoder
  gemm_bf16_wmma<0><<<gN1024, 256, 0, stream>>>(y1_b, Wqab, bqa, Qb, nullptr, MROWS, EMBED, EMBED, QSCALE);
  attn_causal_wmma<<<gAttn, 128, 0, stream>>>(Qb, Keb, Vetb, tmp_f);

  // 7) y2 = LN(ctx2 + y1)
  add_layernorm<<<MROWS, 256, 0, stream>>>(tmp_f, y1_f, g2, be2, y2_f, y2_b);

  // 8) h = gelu(y2 @ W1^T + b1)  (M=4096, N=4096, K=1024)
  dim3 gMLP1(MROWS / 128, FFN / 128);
  gemm_bf16_wmma<1><<<gMLP1, 256, 0, stream>>>(y2_b, W1b, b1, h_b, nullptr, MROWS, FFN, EMBED, 1.0f);

  // 9) mlp = h @ W2^T + b2  (M=4096, N=1024, K=4096) -> f32 tmp
  gemm_bf16_wmma<2><<<gN1024, 256, 0, stream>>>(h_b, W2b, b2, nullptr, tmp_f, MROWS, EMBED, FFN, 1.0f);

  // 10) out = LN(y2 + mlp)
  add_layernorm<<<MROWS, 256, 0, stream>>>(tmp_f, y2_f, g3, be3, (float*)d_out, nullptr);
}
